// MultiHeadAttention_9096740732946
// MI455X (gfx1250) — compile-verified, matrix operand order checked by reading, not run on hardware
//
#include <hip/hip_runtime.h>
#include <hip/hip_bf16.h>

typedef __attribute__((ext_vector_type(16))) __bf16 v16bf;
typedef __attribute__((ext_vector_type(8)))  float  v8f;
typedef float v16f __attribute__((ext_vector_type(16), aligned(16)));
typedef __attribute__((ext_vector_type(4)))  int    v4i;

#define B_  4
#define L_  2048
#define E_  1024
#define H_  16
#define DK_ 64
#define M_  (B_ * L_)   // 8192 tokens

// ---- probe for CDNA5 async global->LDS path (compile-safe) ----------------
#if defined(__has_builtin)
# if __has_builtin(__builtin_amdgcn_global_load_async_to_lds_b128)
#  define USE_ASYNC_LDS 1
# endif
#endif
#ifndef USE_ASYNC_LDS
# define USE_ASYNC_LDS 0
#endif

#if USE_ASYNC_LDS
# if __has_builtin(__builtin_amdgcn_s_wait_asynccnt)
#  define WAIT_ASYNC(n) __builtin_amdgcn_s_wait_asynccnt(n)
# else
#  define WAIT_ASYNC(n) asm volatile("s_wait_asynccnt " #n ::: "memory")
# endif
#else
# define WAIT_ASYNC(n) do {} while (0)
#endif

// ---------------------------------------------------------------------------
// fp32 -> bf16 cast (grid-stride)
// ---------------------------------------------------------------------------
__global__ void cvt_f32_to_bf16(const float* __restrict__ x, __bf16* __restrict__ y, int n)
{
    for (int i = blockIdx.x * blockDim.x + threadIdx.x; i < n; i += gridDim.x * blockDim.x)
        y[i] = (__bf16)x[i];
}

// ---------------------------------------------------------------------------
// C[M,N] = A[M,K](bf16) @ W[N,K](bf16)^T + bias   (torch Linear layout)
// One wave computes a 16x64 tile; block = 8 waves covering 128x64.
// MODE 0: bf16 -> [B,H,L,DK] | MODE 2: bf16 -> [B,H,DK,L] | MODE 3: f32 [M,N]
// ---------------------------------------------------------------------------
template<int MODE>
__global__ void gemm_wmma_bf16(const __bf16* __restrict__ A, const __bf16* __restrict__ W,
                               const float* __restrict__ bias, void* __restrict__ outp,
                               int M, int N, int K)
{
    const int lane  = threadIdx.x & 31;
    const int wave  = threadIdx.x >> 5;
    const int col   = lane & 15;
    const int half  = lane >> 4;
    const int nbase = blockIdx.x * 64;
    const int mbase = (blockIdx.y * 8 + wave) * 16;

    v8f acc[4] = {};
    const __bf16* arow  = A + (size_t)(mbase + col) * K + half * 16;
    const __bf16* wbase = W + (size_t)(nbase + col) * K + half * 16;

    for (int k0 = 0; k0 < K; k0 += 32) {
        v16bf a = *(const v16bf*)(arow + k0);
        __builtin_prefetch(arow + k0 + 128, 0, 0);
#pragma unroll
        for (int t = 0; t < 4; ++t) {
            v16bf b = *(const v16bf*)(wbase + (size_t)t * 16 * K + k0);
            acc[t] = __builtin_amdgcn_wmma_f32_16x16x32_bf16(
                false, a, false, b, (short)0, acc[t], false, false);
        }
    }

#pragma unroll
    for (int t = 0; t < 4; ++t) {
        const int   n  = nbase + t * 16 + col;
        const float bv = bias[n];
#pragma unroll
        for (int r = 0; r < 8; ++r) {
            const int m = mbase + r + half * 8;
            float v = acc[t][r] + bv;
            if (MODE == 3) {
                ((float*)outp)[(size_t)m * N + n] = v;
            } else {
                const int bb = m >> 11, l = m & (L_ - 1);
                const int h = n >> 6, dk = n & (DK_ - 1);
                __bf16* o = (__bf16*)outp;
                if (MODE == 2) o[(((size_t)bb * H_ + h) * DK_ + dk) * L_ + l] = (__bf16)v;
                else           o[(((size_t)bb * H_ + h) * L_ + l) * DK_ + dk] = (__bf16)v;
            }
        }
    }
}

// ---------------------------------------------------------------------------
// Fused flash attention.  Block = 8 waves = 128 query rows of one (b,h).
// K/V 64x64 tiles staged once per block into LDS via async-to-LDS (double
// buffered, ASYNCcnt-tracked).  Softmax done in A-fragment layout: scores
// transposed through a per-wave LDS tile; row stats need only ONE cross-lane
// exchange (lane^16) instead of log2 bpermute chains.
// ---------------------------------------------------------------------------
__global__ void flash_attn_wmma(const __bf16* __restrict__ Q, const __bf16* __restrict__ Kt,
                                const __bf16* __restrict__ Vt, __bf16* __restrict__ ctx,
                                const int* __restrict__ is_causal_p)
{
    const int lane = threadIdx.x & 31;
    const int wave = threadIdx.x >> 5;
    const int colL = lane & 15;
    const int half = lane >> 4;

    const int chunks_per_bh = L_ / 128;              // 16
    const int bh    = blockIdx.x / chunks_per_bh;
    const int chnk  = blockIdx.x % chunks_per_bh;
    const int mb0   = chnk * 128;
    const int mbase = mb0 + wave * 16;
    const int b = bh >> 4, h = bh & 15;
    const int causal = *is_causal_p;

    const __bf16* Qb = Q  + (size_t)bh * L_ * DK_;
    const __bf16* Kb = Kt + (size_t)bh * L_ * DK_;
    const __bf16* Vb = Vt + (size_t)bh * DK_ * L_;

    // staging: K tile [64 keys][64 dk], V tile [64 dk][64 keys] (V pre-transposed)
    __shared__ __align__(16) __bf16 kbuf[2][64][64];
    __shared__ __align__(16) __bf16 vbuf[2][64][64];
    __shared__ __align__(16) float  sbuf[8][16][68];   // per-wave score transpose (+pad)

    auto stage = [&](int bsel, int s) {
        for (int c = (int)threadIdx.x; c < 512; c += 256) {
            const int row = c >> 3, sub = (c & 7) * 8;
            const __bf16* gk = Kb + (size_t)(s + row) * DK_ + sub;
            const __bf16* gv = Vb + (size_t)row * L_ + s + sub;
#if USE_ASYNC_LDS
            __builtin_amdgcn_global_load_async_to_lds_b128(
                (v4i*)gk, (v4i*)&kbuf[bsel][row][sub], 0, 0);
            __builtin_amdgcn_global_load_async_to_lds_b128(
                (v4i*)gv, (v4i*)&vbuf[bsel][row][sub], 0, 0);
#else
            *(v4i*)&kbuf[bsel][row][sub] = *(const v4i*)gk;
            *(v4i*)&vbuf[bsel][row][sub] = *(const v4i*)gv;
#endif
        }
    };

    // Q A-fragments (held in registers for the whole loop)
    const v16bf qa0 = *(const v16bf*)(Qb + (size_t)(mbase + colL) * DK_ + half * 16);
    const v16bf qa1 = *(const v16bf*)(Qb + (size_t)(mbase + colL) * DK_ + 32 + half * 16);

    v8f acc[4] = {};
    float m_st = -3.0e38f, l_st = 0.0f;     // stats of row `colL` (dup across halves)
    const float scale = 0.125f;             // 1/sqrt(64)

    const int my_send    = causal ? ((mbase + 16 + 63) & ~63) : L_;
    const int block_send = causal ? (mb0 + 128) : L_;
    const int nsteps     = block_send >> 6;

    stage(0, 0);

    for (int si = 0; si < nsteps; ++si) {
        const int buf = si & 1;
        const int s   = si << 6;
        if (si + 1 < nsteps) { stage(1 - buf, s + 64); WAIT_ASYNC(4); }
        else                 { WAIT_ASYNC(0); }
        __syncthreads();                       // tile `buf` ready for everyone

        if (s < my_send) {
            // ---- scores S[16 x 64] : 8 WMMAs --------------------------------
            v8f sc[4];
#pragma unroll
            for (int j = 0; j < 4; ++j) {
                v16bf kb0 = *(const v16bf*)&kbuf[buf][j * 16 + colL][half * 16];
                v16bf kb1 = *(const v16bf*)&kbuf[buf][j * 16 + colL][32 + half * 16];
                v8f t = {};
                t = __builtin_amdgcn_wmma_f32_16x16x32_bf16(false, qa0, false, kb0, (short)0, t, false, false);
                t = __builtin_amdgcn_wmma_f32_16x16x32_bf16(false, qa1, false, kb1, (short)0, t, false, false);
                sc[j] = t;
            }

            // ---- scale + causal mask, transpose C-layout -> A-layout via LDS
#pragma unroll
            for (int j = 0; j < 4; ++j) {
                const int n = s + j * 16 + colL;
#pragma unroll
                for (int r = 0; r < 8; ++r) {
                    const int m = mbase + r + 8 * half;
                    float e = sc[j][r] * scale;
                    if (causal && n > m) e = -3.0e38f;
                    sbuf[wave][r + 8 * half][j * 16 + colL] = e;
                }
            }
            asm volatile("" ::: "memory");       // in-wave DS ordering is HW-guaranteed
            v16f f0 = *(const v16f*)&sbuf[wave][colL][half * 16];
            v16f f1 = *(const v16f*)&sbuf[wave][colL][32 + half * 16];
            asm volatile("" ::: "memory");

            // ---- row max: 31 in-register maxes + ONE lane^16 exchange -------
            float mx = fmaxf(f0[0], f1[0]);
#pragma unroll
            for (int i = 1; i < 16; ++i) mx = fmaxf(mx, fmaxf(f0[i], f1[i]));
            mx = fmaxf(mx, __shfl_xor(mx, 16));
            const float nm   = fmaxf(m_st, mx);
            const float corr = __expf(m_st - nm);

            // ---- exp + row sum (values stay in A-fragment order) ------------
            float sum = 0.0f;
            v16f p0, p1;
#pragma unroll
            for (int i = 0; i < 16; ++i) {
                p0[i] = __expf(f0[i] - nm); sum += p0[i];
                p1[i] = __expf(f1[i] - nm); sum += p1[i];
            }
            sum += __shfl_xor(sum, 16);
            l_st = l_st * corr + sum;
            m_st = nm;

            // ---- rescale accumulators (per-row corr broadcast) --------------
#pragma unroll
            for (int r = 0; r < 8; ++r) {
                const float cr = __shfl(corr, r + 8 * half, 32);
#pragma unroll
                for (int t = 0; t < 4; ++t) acc[t][r] *= cr;
            }

            // ---- P already in A-fragment layout: just narrow to bf16 --------
            v16bf pa0, pa1;
#pragma unroll
            for (int i = 0; i < 16; ++i) { pa0[i] = (__bf16)p0[i]; pa1[i] = (__bf16)p1[i]; }

            // ---- ctx += P @ V : 8 WMMAs -------------------------------------
#pragma unroll
            for (int t = 0; t < 4; ++t) {
                v16bf vb0 = *(const v16bf*)&vbuf[buf][t * 16 + colL][half * 16];
                v16bf vb1 = *(const v16bf*)&vbuf[buf][t * 16 + colL][32 + half * 16];
                acc[t] = __builtin_amdgcn_wmma_f32_16x16x32_bf16(false, pa0, false, vb0, (short)0, acc[t], false, false);
                acc[t] = __builtin_amdgcn_wmma_f32_16x16x32_bf16(false, pa1, false, vb1, (short)0, acc[t], false, false);
            }
        }
        __syncthreads();                       // everyone done with tile `buf`
    }

    // ---- normalize, clear NaN, store ctx [B*L, E] bf16 ----------------------
#pragma unroll
    for (int r = 0; r < 8; ++r) {
        const float lr  = __shfl(l_st, r + 8 * half, 32);
        const float inv = 1.0f / lr;
        const int  mrow = mbase + r + half * 8;
#pragma unroll
        for (int t = 0; t < 4; ++t) {
            float v = acc[t][r] * inv;
            v = (v == v) ? v : 0.0f;           // clear_nan
            const int e = h * DK_ + t * 16 + colL;
            ctx[(size_t)(b * L_ + mrow) * E_ + e] = (__bf16)v;
        }
    }
}

// ---------------------------------------------------------------------------
extern "C" void kernel_launch(void* const* d_in, const int* in_sizes, int n_in,
                              void* d_out, int out_size, void* d_ws, size_t ws_size,
                              hipStream_t stream)
{
    const float* query = (const float*)d_in[0];
    const float* keyx  = (const float*)d_in[1];
    const float* value = (const float*)d_in[2];
    const float* wq    = (const float*)d_in[3];
    const float* bq    = (const float*)d_in[4];
    const float* wk    = (const float*)d_in[5];
    const float* bk    = (const float*)d_in[6];
    const float* wv    = (const float*)d_in[7];
    const float* bv    = (const float*)d_in[8];
    const float* wo    = (const float*)d_in[9];
    const float* bo    = (const float*)d_in[10];
    const int*   is_causal = (const int*)d_in[11];
    float* out = (float*)d_out;

    char* w = (char*)d_ws;
    const size_t xsz = (size_t)M_ * E_ * sizeof(__bf16);   // 16.78 MB
    const size_t wsz = (size_t)E_ * E_ * sizeof(__bf16);   //  2 MB
    __bf16* xq   = (__bf16*)(w);
    __bf16* xk   = (__bf16*)(w + xsz);
    __bf16* xv   = (__bf16*)(w + 2 * xsz);
    __bf16* wqb  = (__bf16*)(w + 3 * xsz);
    __bf16* wkb  = (__bf16*)(w + 3 * xsz + wsz);
    __bf16* wvb  = (__bf16*)(w + 3 * xsz + 2 * wsz);
    __bf16* wob  = (__bf16*)(w + 3 * xsz + 3 * wsz);
    __bf16* Qb   = (__bf16*)(w + 3 * xsz + 4 * wsz);
    __bf16* Kb   = (__bf16*)(w + 4 * xsz + 4 * wsz);
    __bf16* Vtb  = (__bf16*)(w + 5 * xsz + 4 * wsz);
    __bf16* ctxb = (__bf16*)(w + 6 * xsz + 4 * wsz);

    const int nx = M_ * E_, nw = E_ * E_;
    cvt_f32_to_bf16<<<2048, 256, 0, stream>>>(query, xq,  nx);
    cvt_f32_to_bf16<<<2048, 256, 0, stream>>>(keyx,  xk,  nx);
    cvt_f32_to_bf16<<<2048, 256, 0, stream>>>(value, xv,  nx);
    cvt_f32_to_bf16<<<512,  256, 0, stream>>>(wq,    wqb, nw);
    cvt_f32_to_bf16<<<512,  256, 0, stream>>>(wk,    wkb, nw);
    cvt_f32_to_bf16<<<512,  256, 0, stream>>>(wv,    wvb, nw);
    cvt_f32_to_bf16<<<512,  256, 0, stream>>>(wo,    wob, nw);

    dim3 gg(E_ / 64, M_ / 128);   // (16, 64) blocks of 256 threads
    gemm_wmma_bf16<0><<<gg, 256, 0, stream>>>(xq, wqb, bq, (void*)Qb,  M_, E_, E_);
    gemm_wmma_bf16<0><<<gg, 256, 0, stream>>>(xk, wkb, bk, (void*)Kb,  M_, E_, E_);
    gemm_wmma_bf16<2><<<gg, 256, 0, stream>>>(xv, wvb, bv, (void*)Vtb, M_, E_, E_);

    flash_attn_wmma<<<B_ * H_ * (L_ / 128), 256, 0, stream>>>(Qb, Kb, Vtb, ctxb, is_causal);

    gemm_wmma_bf16<3><<<gg, 256, 0, stream>>>(ctxb, wob, bo, (void*)out, M_, E_, E_);
}
